// NattenDiTBlock_89661737271384
// MI455X (gfx1250) — matrix operand order repaired and checked by reading, not compiled, not run on hardware
//
#include <hip/hip_runtime.h>

// ---------------------------------------------------------------------------
// NATTEN DiT block for MI455X (gfx1250, wave32, WMMA).
// GEMMs run as bf16 WMMA (v_wmma_f32_16x16x32_bf16) with f32 accumulation;
// double-buffered LDS B-tiles (ping-pong by 2x loop unroll, no register
// copies), 32x64 output strip per wave (8 wmma / 32-K chunk).
// LN / softmax / modulation stay in f32. Intermediates live in d_ws.
// ---------------------------------------------------------------------------

#define HH 64
#define WW 64
#define CC 384
#define NHEAD 6
#define DHEAD 64
#define NPIX (HH * WW)   // 4096
#define KWIN 7
#define KK2 49

typedef __attribute__((ext_vector_type(16))) __bf16 v16bf;
typedef __attribute__((ext_vector_type(8)))  __bf16 v8bf;
typedef __attribute__((ext_vector_type(8)))  float  v8f;
typedef __attribute__((ext_vector_type(8)))  unsigned short u16x8;

__device__ __forceinline__ unsigned short f32_to_bf16(float f) {
    union { float f; unsigned u; } x; x.f = f;
    unsigned r = x.u + 0x7FFFu + ((x.u >> 16) & 1u);   // round-to-nearest-even
    return (unsigned short)(r >> 16);
}
__device__ __forceinline__ float bf16_to_f32(unsigned short h) {
    union { unsigned u; float f; } x; x.u = ((unsigned)h) << 16;
    return x.f;
}

// ---------------------------------------------------------------------------
// f32 -> bf16 elementwise (weight conversion)
// ---------------------------------------------------------------------------
__global__ __launch_bounds__(256) void cvt_bf16(const float* __restrict__ in,
                                                unsigned short* __restrict__ out,
                                                int n) {
    int i = blockIdx.x * 256 + threadIdx.x;
    if (i < n) out[i] = f32_to_bf16(in[i]);
}

// ---------------------------------------------------------------------------
// m = silu(cond) @ ada_w + ada_b      (1x384) @ (384x2304)
// ---------------------------------------------------------------------------
__global__ __launch_bounds__(256) void ada_kernel(const float* __restrict__ cond,
                                                  const float* __restrict__ aw,
                                                  const float* __restrict__ ab,
                                                  float* __restrict__ m) {
    __shared__ float ss[CC];
    for (int i = threadIdx.x; i < CC; i += 256) {
        float c = cond[i];
        ss[i] = c / (1.0f + __expf(-c));
    }
    __syncthreads();
    int col = blockIdx.x * 256 + threadIdx.x;   // grid = 9 -> 2304 cols
    float acc = ab[col];
    for (int k = 0; k < CC; ++k)
        acc = fmaf(ss[k], aw[(size_t)k * (6 * CC) + col], acc);
    m[col] = acc;
}

// ---------------------------------------------------------------------------
// y = ln(x) * (1 + sc) + sh   -> bf16.  One block (128 thr) per row of 384.
// ---------------------------------------------------------------------------
__global__ __launch_bounds__(128) void ln_mod_kernel(const float* __restrict__ x,
                                                     const float* __restrict__ sc,
                                                     const float* __restrict__ sh,
                                                     unsigned short* __restrict__ y) {
    __shared__ float r1[128];
    __shared__ float r2[128];
    const int row = blockIdx.x;
    const int t = threadIdx.x;
    const float* xr = x + (size_t)row * CC;
    float v0 = xr[t], v1 = xr[t + 128], v2 = xr[t + 256];
    r1[t] = v0 + v1 + v2;
    r2[t] = v0 * v0 + v1 * v1 + v2 * v2;
    __syncthreads();
    for (int off = 64; off > 0; off >>= 1) {
        if (t < off) { r1[t] += r1[t + off]; r2[t] += r2[t + off]; }
        __syncthreads();
    }
    const float mu   = r1[0] * (1.0f / CC);
    const float var  = r2[0] * (1.0f / CC) - mu * mu;
    const float rstd = rsqrtf(var + 1e-6f);
    unsigned short* yr = y + (size_t)row * CC;
    yr[t]       = f32_to_bf16((v0 - mu) * rstd * (1.0f + sc[t])       + sh[t]);
    yr[t + 128] = f32_to_bf16((v1 - mu) * rstd * (1.0f + sc[t + 128]) + sh[t + 128]);
    yr[t + 256] = f32_to_bf16((v2 - mu) * rstd * (1.0f + sc[t + 256]) + sh[t + 256]);
}

// ---------------------------------------------------------------------------
// Tiled bf16 WMMA GEMM:  out[M,N] = epi(A[M,K] @ B[K,N] + bias)
// Block tile 256(M) x 64(N), 256 threads = 8 waves; each wave owns a 32x64
// strip (2 A fragments x 4 B fragments = 8 wmma per 32-K chunk).
// B chunks (32x64) double-buffered in LDS (transposed, so fragment reads are
// 2x ds_load_b128). The K loop is unrolled by 2 with two A-fragment register
// sets so the ping-pong needs no register moves and LDS offsets are
// compile-time. Next-chunk global loads issue before the wmma burst.
// EPI 0: bf16 store, columns < scaleCols multiplied by scaleVal (q scaling)
// EPI 1: f32 store, out = resid + gate[n] * (acc + bias)
// EPI 2: bf16 store, tanh-GELU(acc + bias)
// ---------------------------------------------------------------------------
template <int EPI>
__global__ __launch_bounds__(256) void wmma_gemm(const unsigned short* __restrict__ A,
                                                 const unsigned short* __restrict__ B,
                                                 const float* __restrict__ bias,
                                                 const float* __restrict__ resid,
                                                 const float* __restrict__ gate,
                                                 float* __restrict__ outF,
                                                 unsigned short* __restrict__ outB,
                                                 int N, int K,
                                                 int scaleCols, float scaleVal) {
    __shared__ unsigned short Bs[2][64 * 40];   // [buf][n_local][k], pad 40

    const int tid   = threadIdx.x;
    const int lane  = tid & 31;
    const int wave  = tid >> 5;
    const int half  = lane >> 4;          // K-half select for fragments
    const int l15   = lane & 15;
    const int nblk  = blockIdx.x * 64;
    const int mbase = blockIdx.y * 256 + wave * 32;

    // staging thread mapping: k = tid>>3 (0..31), ng = tid&7 (8-col group)
    const int sk  = tid >> 3;
    const int sng = tid & 7;
    const unsigned short* bsrc = B + (size_t)sk * N + nblk + sng * 8;

    // A row pointers for the wave's two 16-row fragments (lane -> row l15)
    const unsigned short* arow0 = A + (size_t)(mbase + l15) * K + half * 8;
    const unsigned short* arow1 = arow0 + (size_t)16 * K;

    v8f acc[2][4];
    #pragma unroll
    for (int ms = 0; ms < 2; ++ms)
        #pragma unroll
        for (int ns = 0; ns < 4; ++ns) acc[ms][ns] = v8f{};

    const int kchunks = K >> 5;   // always even here (12 or 48)

    // one pipeline step: compute chunk from bsCur/apc, prefetch knext into
    // bsNxt/apn (roles swap by call position -> no register moves)
    auto step = [&](unsigned short* bsCur, unsigned short* bsNxt,
                    v8bf (&apc)[2][2], v8bf (&apn)[2][2],
                    int knext, bool hasNext) {
        // issue next-chunk global loads first (overlap with wmma burst)
        u16x8 bnext{};
        if (hasNext) {
            bnext = *(const u16x8*)(bsrc + (size_t)knext * N);
            apn[0][0] = *(const v8bf*)(arow0 + knext);
            apn[0][1] = *(const v8bf*)(arow0 + knext + 16);
            apn[1][0] = *(const v8bf*)(arow1 + knext);
            apn[1][1] = *(const v8bf*)(arow1 + knext + 16);
        }

        // build the two A fragments
        v16bf af[2];
        #pragma unroll
        for (int ms = 0; ms < 2; ++ms)
            #pragma unroll
            for (int i = 0; i < 8; ++i) {
                af[ms][i]     = apc[ms][0][i];
                af[ms][i + 8] = apc[ms][1][i];
            }

        // 8 wmma: 4 B fragments x 2 A fragments (B reused from registers)
        #pragma unroll
        for (int ns = 0; ns < 4; ++ns) {
            const unsigned short* bp = bsCur + (ns * 16 + l15) * 40 + half * 8;
            v8bf blo = *(const v8bf*)bp;
            v8bf bhi = *(const v8bf*)(bp + 16);
            v16bf bfrag;
            #pragma unroll
            for (int i = 0; i < 8; ++i) { bfrag[i] = blo[i]; bfrag[i + 8] = bhi[i]; }
            acc[0][ns] = __builtin_amdgcn_wmma_f32_16x16x32_bf16(
                false, af[0], false, bfrag, (short)0, acc[0][ns], false, false);
            acc[1][ns] = __builtin_amdgcn_wmma_f32_16x16x32_bf16(
                false, af[1], false, bfrag, (short)0, acc[1][ns], false, false);
        }

        // commit next-chunk staging into the other buffer
        if (hasNext) {
            #pragma unroll
            for (int j = 0; j < 8; ++j)
                bsNxt[(sng * 8 + j) * 40 + sk] = bnext[j];
        }
        __syncthreads();
    };

    // ---- prologue: stage chunk 0 into buffer 0, prefetch A chunk 0 ----
    {
        u16x8 bv = *(const u16x8*)bsrc;
        #pragma unroll
        for (int j = 0; j < 8; ++j)
            Bs[0][(sng * 8 + j) * 40 + sk] = bv[j];
    }
    v8bf apA[2][2], apB[2][2];
    apA[0][0] = *(const v8bf*)(arow0);
    apA[0][1] = *(const v8bf*)(arow0 + 16);
    apA[1][0] = *(const v8bf*)(arow1);
    apA[1][1] = *(const v8bf*)(arow1 + 16);
    __syncthreads();

    for (int kc = 0; kc < kchunks; kc += 2) {
        step(&Bs[0][0], &Bs[1][0], apA, apB, (kc + 1) << 5, true);
        step(&Bs[1][0], &Bs[0][0], apB, apA, (kc + 2) << 5, (kc + 2) < kchunks);
    }

    // Epilogue. C/D layout: lane -> N = l15 ; VGPR r -> M = r + 8*half.
    #pragma unroll
    for (int ms = 0; ms < 2; ++ms) {
        #pragma unroll
        for (int ns = 0; ns < 4; ++ns) {
            const int n = nblk + ns * 16 + l15;
            const float bn = bias[n];
            #pragma unroll
            for (int r = 0; r < 8; ++r) {
                const int m = mbase + ms * 16 + r + 8 * half;
                float v = acc[ms][ns][r] + bn;
                const size_t idx = (size_t)m * N + n;
                if (EPI == 0) {
                    float s = (n < scaleCols) ? scaleVal : 1.0f;
                    outB[idx] = f32_to_bf16(v * s);
                } else if (EPI == 1) {
                    outF[idx] = resid[idx] + gate[n] * v;
                } else {
                    float g = 0.5f * v *
                        (1.0f + tanhf(0.7978845608f * (v + 0.044715f * v * v * v)));
                    outB[idx] = f32_to_bf16(g);
                }
            }
        }
    }
}

// ---------------------------------------------------------------------------
// Neighborhood attention, 7x7 boundary-clamped window.
// One wave per (pixel, head). q cached in LDS; each lane scores 2 neighbors,
// wave-wide shuffle softmax; each lane accumulates 2 of the 64 output dims.
// qkv layout: [pix][ q(384) | k(384) | v(384) ], q pre-scaled by DH^-0.5.
// ---------------------------------------------------------------------------
__global__ __launch_bounds__(256) void natten_kernel(const unsigned short* __restrict__ qkv,
                                                     unsigned short* __restrict__ o) {
    __shared__ float qsh[8][DHEAD];
    __shared__ float psh[8][64];

    const int lane = threadIdx.x & 31;
    const int wave = threadIdx.x >> 5;
    const int task = blockIdx.x * 8 + wave;          // < 4096*6
    const int head = task % NHEAD;
    const int pix  = task / NHEAD;
    const int h = pix >> 6, w = pix & 63;

    int sh0 = h - 3; if (sh0 < 0) sh0 = 0; if (sh0 > HH - KWIN) sh0 = HH - KWIN;
    int sw0 = w - 3; if (sw0 < 0) sw0 = 0; if (sw0 > WW - KWIN) sw0 = WW - KWIN;

    const size_t qb = (size_t)pix * (3 * CC) + head * DHEAD;
    qsh[wave][lane]      = bf16_to_f32(qkv[qb + lane]);
    qsh[wave][lane + 32] = bf16_to_f32(qkv[qb + lane + 32]);
    __syncthreads();

    float s[2] = {-1e30f, -1e30f};
    #pragma unroll
    for (int hf = 0; hf < 2; ++hf) {
        const int j = lane + 32 * hf;
        if (j < KK2) {
            const int kh = sh0 + j / KWIN;
            const int kw = sw0 + j % KWIN;
            const unsigned short* kp =
                qkv + (size_t)(kh * WW + kw) * (3 * CC) + CC + head * DHEAD;
            float d = 0.0f;
            #pragma unroll 8
            for (int t = 0; t < DHEAD; ++t)
                d = fmaf(qsh[wave][t], bf16_to_f32(kp[t]), d);
            s[hf] = d;
        }
    }

    float mx = fmaxf(s[0], s[1]);
    #pragma unroll
    for (int msk = 16; msk >= 1; msk >>= 1)
        mx = fmaxf(mx, __shfl_xor(mx, msk, 32));
    float e0 = __expf(s[0] - mx);
    float e1 = __expf(s[1] - mx);
    float sum = e0 + e1;
    #pragma unroll
    for (int msk = 16; msk >= 1; msk >>= 1)
        sum += __shfl_xor(sum, msk, 32);
    const float inv = 1.0f / sum;
    psh[wave][lane]      = e0 * inv;
    psh[wave][lane + 32] = e1 * inv;
    __syncthreads();

    float a0 = 0.0f, a1 = 0.0f;
    for (int j = 0; j < KK2; ++j) {
        const int kh = sh0 + j / KWIN;
        const int kw = sw0 + j % KWIN;
        const unsigned short* vp =
            qkv + (size_t)(kh * WW + kw) * (3 * CC) + 2 * CC + head * DHEAD;
        const float p = psh[wave][j];
        a0 = fmaf(p, bf16_to_f32(vp[lane]), a0);
        a1 = fmaf(p, bf16_to_f32(vp[lane + 32]), a1);
    }
    unsigned short* orow = o + (size_t)pix * CC + head * DHEAD;
    orow[lane]      = f32_to_bf16(a0);
    orow[lane + 32] = f32_to_bf16(a1);
}

// ---------------------------------------------------------------------------
// Host launcher
// ---------------------------------------------------------------------------
extern "C" void kernel_launch(void* const* d_in, const int* in_sizes, int n_in,
                              void* d_out, int out_size, void* d_ws, size_t ws_size,
                              hipStream_t stream) {
    (void)in_sizes; (void)n_in; (void)out_size; (void)ws_size;

    const float* x      = (const float*)d_in[0];   // [4096,384]
    const float* cond   = (const float*)d_in[1];   // [384]
    const float* qkv_w  = (const float*)d_in[2];   // [384,1152]
    const float* qkv_b  = (const float*)d_in[3];
    const float* proj_w = (const float*)d_in[4];   // [384,384]
    const float* proj_b = (const float*)d_in[5];
    const float* fc1_w  = (const float*)d_in[6];   // [384,1536]
    const float* fc1_b  = (const float*)d_in[7];
    const float* fc2_w  = (const float*)d_in[8];   // [1536,384]
    const float* fc2_b  = (const float*)d_in[9];
    const float* ada_w  = (const float*)d_in[10];  // [384,2304]
    const float* ada_b  = (const float*)d_in[11];
    float* out = (float*)d_out;

    char* ws = (char*)d_ws;
    size_t off = 0;
    auto take = [&](size_t bytes) {
        size_t o = off;
        off += (bytes + 255) & ~(size_t)255;
        return o;
    };
    float*          mbuf  = (float*)(ws + take((size_t)6 * CC * 4));
    unsigned short* y1    = (unsigned short*)(ws + take((size_t)NPIX * CC * 2));
    unsigned short* qkvB  = (unsigned short*)(ws + take((size_t)NPIX * 3 * CC * 2));
    unsigned short* oB    = (unsigned short*)(ws + take((size_t)NPIX * CC * 2));
    float*          x1    = (float*)(ws + take((size_t)NPIX * CC * 4));
    unsigned short* y2    = (unsigned short*)(ws + take((size_t)NPIX * CC * 2));
    unsigned short* hdn   = (unsigned short*)(ws + take((size_t)NPIX * 4 * CC * 2));
    unsigned short* qkvW  = (unsigned short*)(ws + take((size_t)CC * 3 * CC * 2));
    unsigned short* projW = (unsigned short*)(ws + take((size_t)CC * CC * 2));
    unsigned short* fc1W  = (unsigned short*)(ws + take((size_t)CC * 4 * CC * 2));
    unsigned short* fc2W  = (unsigned short*)(ws + take((size_t)4 * CC * CC * 2));

    // Weight conversion f32 -> bf16
    auto cvt = [&](const float* src, unsigned short* dst, int n) {
        cvt_bf16<<<(n + 255) / 256, 256, 0, stream>>>(src, dst, n);
    };
    cvt(qkv_w,  qkvW,  CC * 3 * CC);
    cvt(proj_w, projW, CC * CC);
    cvt(fc1_w,  fc1W,  CC * 4 * CC);
    cvt(fc2_w,  fc2W,  4 * CC * CC);

    // Modulation: m = silu(cond) @ ada_w + ada_b  (6*384 = 2304 = 9*256)
    ada_kernel<<<9, 256, 0, stream>>>(cond, ada_w, ada_b, mbuf);
    const float* sh_a = mbuf + 0 * CC;
    const float* sc_a = mbuf + 1 * CC;
    const float* g_a  = mbuf + 2 * CC;
    const float* sh_m = mbuf + 3 * CC;
    const float* sc_m = mbuf + 4 * CC;
    const float* g_m  = mbuf + 5 * CC;

    // Attention branch
    ln_mod_kernel<<<NPIX, 128, 0, stream>>>(x, sc_a, sh_a, y1);
    // qkv: [4096,384] @ [384,1152]; fuse q * DH^-0.5 into the store
    wmma_gemm<0><<<dim3(3 * CC / 64, NPIX / 256), 256, 0, stream>>>(
        y1, qkvW, qkv_b, nullptr, nullptr, nullptr, qkvB,
        3 * CC, CC, CC, 0.125f);
    natten_kernel<<<NPIX * NHEAD / 8, 256, 0, stream>>>(qkvB, oB);
    // proj + residual: x1 = x + g_a * (o @ proj_w + proj_b)
    wmma_gemm<1><<<dim3(CC / 64, NPIX / 256), 256, 0, stream>>>(
        oB, projW, proj_b, x, g_a, x1, nullptr,
        CC, CC, 0, 1.0f);

    // MLP branch
    ln_mod_kernel<<<NPIX, 128, 0, stream>>>(x1, sc_m, sh_m, y2);
    // fc1 + GELU: hdn = gelu(y2 @ fc1_w + fc1_b)
    wmma_gemm<2><<<dim3(4 * CC / 64, NPIX / 256), 256, 0, stream>>>(
        y2, fc1W, fc1_b, nullptr, nullptr, nullptr, hdn,
        4 * CC, CC, 0, 1.0f);
    // fc2 + residual -> d_out
    wmma_gemm<1><<<dim3(CC / 64, NPIX / 256), 256, 0, stream>>>(
        hdn, fc2W, fc2_b, x1, g_m, out, nullptr,
        CC, 4 * CC, 0, 1.0f);
}
